// CrossModalityCrossAttention_22668837388655
// MI455X (gfx1250) — matrix operand order repaired and checked by reading, not compiled, not run on hardware
//
#include <hip/hip_runtime.h>
#include <hip/hip_bf16.h>
#include <math.h>

typedef _Float16 f16;
typedef __attribute__((ext_vector_type(16))) _Float16 v16h;
typedef __attribute__((ext_vector_type(8)))  float    v8f;
typedef unsigned u32;
typedef __attribute__((ext_vector_type(4))) unsigned u32x4;
typedef __attribute__((ext_vector_type(8))) unsigned u32x8;

#define HEADS   8
#define DH      64
#define NCHUNK  64
#define CHUNKQ  64
#define JP      160      // 129 keys padded to 5 * 32
#define BATCH   4
#define SEQ     4097
#define CTXLEN  8065
#define DMODEL  1024
#define INNER   512

// ---------------------------------------------------------------------------
// WMMA helpers (CDNA5 v_wmma_f32_16x16x32_f16, wave32)
// ---------------------------------------------------------------------------
__device__ __forceinline__ v8f wmma_f16(v16h a, v16h b, v8f c) {
  return __builtin_amdgcn_wmma_f32_16x16x32_f16(false, a, false, b, (short)0, c,
                                                false, false);
}

// A fragment: 16x32 f16, LDS row-major [m][k], row stride ld (halves), k offset koff.
__device__ __forceinline__ v16h load_frag_a(const f16* p0, int row0, int ld, int koff) {
  const int lane = threadIdx.x & 31;
  const int hi = (threadIdx.x >> 4) & 1;
  const unsigned* p = (const unsigned*)(p0 + (size_t)(row0 + (lane & 15)) * ld + koff);
  union { v16h v; unsigned u[8]; } r;
#pragma unroll
  for (int i = 0; i < 4; ++i) r.u[i] = p[(hi * 8 + 2 * i) >> 1];
#pragma unroll
  for (int i = 0; i < 4; ++i) r.u[4 + i] = p[(16 + hi * 8 + 2 * i) >> 1];
  return r.v;
}

// B fragment: 32x16 f16, LDS stored transposed [n][k], row stride ld, k offset koff.
__device__ __forceinline__ v16h load_frag_b(const f16* p0, int n0, int ld, int koff) {
  const int lane = threadIdx.x & 31;
  const int kb = ((threadIdx.x >> 4) & 1) * 8 + koff;
  const unsigned* p = (const unsigned*)(p0 + (size_t)(n0 + (lane & 15)) * ld);
  union { v16h v; unsigned u[8]; } r;
#pragma unroll
  for (int i = 0; i < 8; ++i) r.u[i] = p[(kb + (i >> 2) * 16 + 2 * (i & 3)) >> 1];
  return r.v;
}

// ---------------------------------------------------------------------------
// CDNA5 async data movers
// ---------------------------------------------------------------------------
__device__ __forceinline__ unsigned lds_offset(const void* p) {
  // LDS flat address low 32 bits == wave LDS offset (ISA 10.2 aperture rules)
  return (unsigned)(size_t)p;
}

__device__ __forceinline__ void async_load_b128(unsigned ldsoff, const void* g) {
  asm volatile("global_load_async_to_lds_b128 %0, %1, off"
               :: "v"(ldsoff), "v"((unsigned long long)(size_t)g)
               : "memory");
}

__device__ __forceinline__ void wait_asynccnt0() {
  asm volatile("s_wait_asynccnt 0x0" ::: "memory");
}

// TDM 2D tile load: rows x elems of 8-byte elements, contiguous (stride == elems).
// D# layout per CDNA5 ISA.
__device__ __forceinline__ void tdm_load_2d_b8e(unsigned ldsoff, const void* gaddr,
                                                u32 elems, u32 rows) {
  unsigned long long ga = (unsigned long long)(size_t)gaddr;
  u32x4 g0;
  g0.x = 1u;                                        // count=1, user mode
  g0.y = ldsoff;                                    // lds_addr
  g0.z = (u32)ga;                                   // global_addr[31:0]
  g0.w = (u32)(ga >> 32) | 0x80000000u;             // global_addr[56:32] | type=2
  u32x8 g1;
  g1.s0 = 0x00030000u;                              // data_size=3 (8B), mask=0
  g1.s1 = (elems & 0xffffu) << 16;                  // tensor_dim0[15:0] @ bits63:48
  g1.s2 = ((rows & 0xffffu) << 16) | (elems >> 16); // tensor_dim1[15:0] | td0[31:16]
  g1.s3 = ((elems & 0xffffu) << 16) | (rows >> 16); // tile_dim0 | td1[31:16]
  g1.s4 = rows & 0xffffu;                           // tile_dim1 (tile_dim2=0)
  g1.s5 = elems;                                    // tensor_dim0_stride[31:0]
  g1.s6 = 0u;                                       // stride0[47:32] | stride1[15:0]
  g1.s7 = 0u;                                       // stride1[47:16]
  asm volatile("tensor_load_to_lds %0, %1" :: "s"(g0), "s"(g1) : "memory");
}

// ---------------------------------------------------------------------------
// Kernel 1: Q = s @ Wq   (M=16384, N=512, K=1024), scaled by DIM_HEAD^-0.5
// ---------------------------------------------------------------------------
__global__ void __launch_bounds__(128) proj_q_kernel(const float* __restrict__ seq,
                                                     const float* __restrict__ Wq,
                                                     f16* __restrict__ Qf) {
  __shared__ __align__(16) f16 As[64 * 32];
  __shared__ __align__(16) f16 Bs[64 * 32];  // [n][k]
  const int tilesN = INNER / 64;  // 8
  const int tM = blockIdx.x / tilesN;
  const int tN = blockIdx.x % tilesN;
  const int m0 = tM * 64, n0 = tN * 64;
  const int b = m0 >> 12;
  const int r0 = m0 & 4095;
  const int tid = threadIdx.x, wave = tid >> 5, lane = tid & 31;
  v8f acc[4] = {};
  for (int kt = 0; kt < DMODEL; kt += 32) {
    if (kt + 32 < DMODEL)
      __builtin_prefetch(&seq[((size_t)b * SEQ + (r0 + 1)) * DMODEL + kt + 32], 0, 1);
    for (int idx = tid; idx < 64 * 32; idx += 128) {
      int r = idx >> 5, c = idx & 31;
      As[r * 32 + c] = (f16)seq[((size_t)b * SEQ + (r0 + r + 1)) * DMODEL + kt + c];
    }
    for (int idx = tid; idx < 32 * 64; idx += 128) {
      int k = idx >> 6, n = idx & 63;
      Bs[n * 32 + k] = (f16)Wq[(size_t)(kt + k) * INNER + n0 + n];
    }
    __syncthreads();
    v16h fa = load_frag_a(As, wave * 16, 32, 0);
#pragma unroll
    for (int nt = 0; nt < 4; ++nt) {
      v16h fb = load_frag_b(Bs, nt * 16, 32, 0);
      acc[nt] = wmma_f16(fa, fb, acc[nt]);
    }
    __syncthreads();
  }
#pragma unroll
  for (int nt = 0; nt < 4; ++nt) {
    int e = n0 + nt * 16 + (lane & 15);
    int h = e >> 6, dd = e & 63;
#pragma unroll
    for (int r = 0; r < 8; ++r) {
      int mrow = r0 + wave * 16 + ((lane >> 4) << 3) + r;
      int chunk = mrow >> 6, i = mrow & 63;
      size_t qi = ((((size_t)b * HEADS + h) * NCHUNK + chunk) * CHUNKQ + i) * DH + dd;
      Qf[qi] = (f16)(0.125f * acc[nt][r]);
    }
  }
}

// ---------------------------------------------------------------------------
// Kernel 2: KV = c_padded @ Wkv  (M=32768, N=1024, K=1024)
// ---------------------------------------------------------------------------
__global__ void __launch_bounds__(128) proj_kv_kernel(const float* __restrict__ ctx,
                                                      const float* __restrict__ Wkv,
                                                      f16* __restrict__ Kf,
                                                      f16* __restrict__ Vf) {
  __shared__ __align__(16) f16 As[64 * 32];
  __shared__ __align__(16) f16 Bs[64 * 32];
  const int tilesN = DMODEL / 64;  // 16
  const int tM = blockIdx.x / tilesN;
  const int tN = blockIdx.x % tilesN;
  const int m0 = tM * 64, n0 = tN * 64;
  const int b = m0 >> 13;
  const int p0 = m0 & 8191;
  const int tid = threadIdx.x, wave = tid >> 5, lane = tid & 31;
  v8f acc[4] = {};
  for (int kt = 0; kt < DMODEL; kt += 32) {
    for (int idx = tid; idx < 64 * 32; idx += 128) {
      int r = idx >> 5, c = idx & 31;
      int p = p0 + r;
      float v = (p >= 127) ? ctx[((size_t)b * CTXLEN + (p - 127)) * DMODEL + kt + c] : 0.f;
      As[r * 32 + c] = (f16)v;
    }
    for (int idx = tid; idx < 32 * 64; idx += 128) {
      int k = idx >> 6, n = idx & 63;
      Bs[n * 32 + k] = (f16)Wkv[(size_t)(kt + k) * DMODEL + n0 + n];
    }
    __syncthreads();
    v16h fa = load_frag_a(As, wave * 16, 32, 0);
#pragma unroll
    for (int nt = 0; nt < 4; ++nt) {
      v16h fb = load_frag_b(Bs, nt * 16, 32, 0);
      acc[nt] = wmma_f16(fa, fb, acc[nt]);
    }
    __syncthreads();
  }
#pragma unroll
  for (int nt = 0; nt < 4; ++nt) {
    int e = n0 + nt * 16 + (lane & 15);
#pragma unroll
    for (int r = 0; r < 8; ++r) {
      int p = p0 + wave * 16 + ((lane >> 4) << 3) + r;
      int chunk = p >> 7, j = (p & 127) + 1;
      f16* dst; int h, dd = e & 63;
      if (e < INNER) { h = e >> 6; dst = Kf; }
      else           { h = (e - INNER) >> 6; dst = Vf; }
      size_t oi = ((((size_t)b * HEADS + h) * NCHUNK + chunk) * JP + j) * DH + dd;
      dst[oi] = (f16)acc[nt][r];
    }
  }
}

// ---------------------------------------------------------------------------
// Kernel 2b: fill null_k/null_v at j=0 and zero the pad rows j=129..159
// ---------------------------------------------------------------------------
__global__ void fill_null_pad_kernel(const float* __restrict__ null_k,
                                     const float* __restrict__ null_v,
                                     f16* __restrict__ Kf, f16* __restrict__ Vf) {
  const int bhc = blockIdx.x;
  const int t = threadIdx.x;  // 64 = d
  const int h = (bhc / NCHUNK) % HEADS;
  const size_t base = (size_t)bhc * JP * DH;
  Kf[base + t] = (f16)null_k[h * DH + t];
  Vf[base + t] = (f16)null_v[h * DH + t];
  for (int j = 129; j < JP; ++j) {
    Kf[base + (size_t)j * DH + t] = (f16)0.f;
    Vf[base + (size_t)j * DH + t] = (f16)0.f;
  }
}

// ---------------------------------------------------------------------------
// Kernel 3: per (b,h,chunk): sim = Q @ K^T, softmax -> Attn f16.
// Q tile via global_load_async_to_lds_b128; K tile via TDM tensor_load_to_lds.
// ---------------------------------------------------------------------------
__global__ void __launch_bounds__(128) attn_scores_kernel(const f16* __restrict__ Qf,
                                                          const f16* __restrict__ Kf,
                                                          f16* __restrict__ Attn) {
  __shared__ __align__(16) unsigned char smem[64 * JP * 4];  // 40 KB
  f16* Qs = (f16*)smem;                    // 64x64   (8 KB)
  f16* Ks = (f16*)(smem + 64 * 64 * 2);    // 160x64  (20 KB) -> [j][d]
  float* Sim = (float*)smem;               // reused: 64x160 f32
  const int bhc = blockIdx.x;
  const int tid = threadIdx.x, wave = tid >> 5, lane = tid & 31;

  // K tile: one TDM descriptor moves the whole 160x64 f16 tile (160 rows x 16 x 8B)
  if (wave == 0) {
    tdm_load_2d_b8e(lds_offset(Ks), Kf + (size_t)bhc * JP * 64, 16u, (u32)JP);
    __builtin_amdgcn_s_wait_tensorcnt((unsigned short)0);
  }
  // Q tile: async global->LDS, 512 x 16B
  {
    const char* qsrc = (const char*)(Qf + (size_t)bhc * 64 * 64);
    unsigned qlds = lds_offset(Qs);
    for (int i = tid; i < (64 * 64 * 2) / 16; i += 128)
      async_load_b128(qlds + i * 16, qsrc + (size_t)i * 16);
    wait_asynccnt0();
  }
  __syncthreads();

  v16h fa0 = load_frag_a(Qs, wave * 16, 64, 0);
  v16h fa1 = load_frag_a(Qs, wave * 16, 64, 32);
  v8f acc[10];
#pragma unroll
  for (int jt = 0; jt < 10; ++jt) {
    v16h fb0 = load_frag_b(Ks, jt * 16, 64, 0);
    v16h fb1 = load_frag_b(Ks, jt * 16, 64, 32);
    v8f c = {};
    c = wmma_f16(fa0, fb0, c);
    c = wmma_f16(fa1, fb1, c);
    acc[jt] = c;
  }
  __syncthreads();  // done with Qs/Ks, reuse as Sim
#pragma unroll
  for (int jt = 0; jt < 10; ++jt) {
    int n = jt * 16 + (lane & 15);
    int mb = wave * 16 + ((lane >> 4) << 3);
#pragma unroll
    for (int r = 0; r < 8; ++r) Sim[(mb + r) * JP + n] = acc[jt][r];
  }
  __syncthreads();
  if (tid < 64) {
    float* srow = Sim + tid * JP;
    float mx = -1e30f;
    for (int j = 0; j < 129; ++j) mx = fmaxf(mx, srow[j]);
    float sum = 0.f;
    for (int j = 0; j < 129; ++j) sum += __expf(srow[j] - mx);
    float inv = 1.f / sum;
    f16* arow = Attn + (size_t)bhc * 64 * JP + (size_t)tid * JP;
    for (int j = 0; j < 129; ++j) arow[j] = (f16)(__expf(srow[j] - mx) * inv);
    for (int j = 129; j < JP; ++j) arow[j] = (f16)0.f;
  }
}

// ---------------------------------------------------------------------------
// Kernel 4: per (b,g,chunk): mix = sum_h W_th[g,h]*attn_h + b_th[g] (j<129),
// then out = mix(64x160) @ V_g(160x64). Write OutMid f16.
// ---------------------------------------------------------------------------
__global__ void __launch_bounds__(128) attn_mix_av_kernel(const f16* __restrict__ Attn,
                                                          const f16* __restrict__ Vf,
                                                          const float* __restrict__ W_th,
                                                          const float* __restrict__ b_th,
                                                          f16* __restrict__ OutMid) {
  __shared__ __align__(16) f16 Mix[64 * JP];  // [i][j]
  __shared__ __align__(16) f16 Vs[64 * JP];   // [d][j] (transposed)
  const int bgc = blockIdx.x;
  const int chunk = bgc % NCHUNK;
  const int g = (bgc / NCHUNK) % HEADS;
  const int b = bgc / (NCHUNK * HEADS);
  const int tid = threadIdx.x, wave = tid >> 5, lane = tid & 31;

  float wg[8];
#pragma unroll
  for (int h = 0; h < 8; ++h) wg[h] = W_th[g * 8 + h];
  const float bias = b_th[g];

  const size_t hstride = (size_t)NCHUNK * 64 * JP;
  const size_t abase = ((size_t)b * HEADS) * hstride + (size_t)chunk * 64 * JP;
  for (int idx = tid; idx < 64 * JP; idx += 128) {
    int i = idx / JP, j = idx - i * JP;
    float s = (j < 129) ? bias : 0.f;
#pragma unroll
    for (int h = 0; h < 8; ++h)
      s += wg[h] * (float)Attn[abase + (size_t)h * hstride + (size_t)i * JP + j];
    Mix[i * JP + j] = (f16)s;
  }
  const size_t vbase = (((size_t)b * HEADS + g) * NCHUNK + chunk) * (size_t)(JP * DH);
  for (int idx = tid; idx < JP * DH; idx += 128) {
    int j = idx >> 6, d = idx & 63;
    Vs[d * JP + j] = Vf[vbase + idx];
  }
  __syncthreads();

  v16h fa[5];
#pragma unroll
  for (int kt = 0; kt < 5; ++kt) fa[kt] = load_frag_a(Mix, wave * 16, JP, kt * 32);
#pragma unroll
  for (int nt = 0; nt < 4; ++nt) {
    v8f c = {};
#pragma unroll
    for (int kt = 0; kt < 5; ++kt) {
      v16h fb = load_frag_b(Vs, nt * 16, JP, kt * 32);
      c = wmma_f16(fa[kt], fb, c);
    }
    int col = g * 64 + nt * 16 + (lane & 15);
    int mb = wave * 16 + ((lane >> 4) << 3);
#pragma unroll
    for (int r = 0; r < 8; ++r) {
      int row = chunk * 64 + mb + r;
      OutMid[((size_t)b * 4096 + row) * INNER + col] = (f16)c[r];
    }
  }
}

// ---------------------------------------------------------------------------
// Kernel 5: Y = OutMid @ Wout + b_out  (M=16384, N=1024, K=512), shifted +1 row
// A tile staged with async global->LDS (already f16).
// ---------------------------------------------------------------------------
__global__ void __launch_bounds__(128) proj_out_kernel(const f16* __restrict__ OutMid,
                                                       const float* __restrict__ Wout,
                                                       const float* __restrict__ b_out,
                                                       float* __restrict__ Out) {
  __shared__ __align__(16) f16 As[64 * 32];
  __shared__ __align__(16) f16 Bs[64 * 32];
  const int tilesN = DMODEL / 64;  // 16
  const int tM = blockIdx.x / tilesN;
  const int tN = blockIdx.x % tilesN;
  const int m0 = tM * 64, n0 = tN * 64;
  const int tid = threadIdx.x, wave = tid >> 5, lane = tid & 31;
  v8f acc[4] = {};
  const char* abase = (const char*)OutMid;
  const unsigned alds = lds_offset(As);
  for (int kt = 0; kt < INNER; kt += 32) {
    if (kt + 32 < INNER)
      __builtin_prefetch(&OutMid[(size_t)m0 * INNER + kt + 32], 0, 1);
    // A tile: 64 rows x 64B, async global->LDS (2 x 16B per thread)
    for (int idx = tid; idx < 256; idx += 128) {
      int r = idx >> 2, c = idx & 3;
      async_load_b128(alds + (unsigned)(r * 64 + c * 16),
                      abase + ((size_t)(m0 + r) * INNER + kt) * 2 + c * 16);
    }
    for (int idx = tid; idx < 32 * 64; idx += 128) {
      int k = idx >> 6, n = idx & 63;
      Bs[n * 32 + k] = (f16)Wout[(size_t)(kt + k) * DMODEL + n0 + n];
    }
    wait_asynccnt0();
    __syncthreads();
    v16h fa = load_frag_a(As, wave * 16, 32, 0);
#pragma unroll
    for (int nt = 0; nt < 4; ++nt) {
      v16h fb = load_frag_b(Bs, nt * 16, 32, 0);
      acc[nt] = wmma_f16(fa, fb, acc[nt]);
    }
    __syncthreads();
  }
#pragma unroll
  for (int nt = 0; nt < 4; ++nt) {
    int col = n0 + nt * 16 + (lane & 15);
    float bo = b_out[col];
    int mb = wave * 16 + ((lane >> 4) << 3);
#pragma unroll
    for (int r = 0; r < 8; ++r) {
      int m = m0 + mb + r;
      int b = m >> 12, row = m & 4095;
      Out[((size_t)b * SEQ + row + 1) * DMODEL + col] = acc[nt][r] + bo;
    }
  }
}

__global__ void zero_first_row_kernel(float* __restrict__ Out) {
  Out[(size_t)blockIdx.x * SEQ * DMODEL + threadIdx.x] = 0.f;
}

// ---------------------------------------------------------------------------
extern "C" void kernel_launch(void* const* d_in, const int* in_sizes, int n_in,
                              void* d_out, int out_size, void* d_ws, size_t ws_size,
                              hipStream_t stream) {
  const float* seq    = (const float*)d_in[0];
  const float* ctx    = (const float*)d_in[1];
  const float* Wq     = (const float*)d_in[2];
  const float* Wkv    = (const float*)d_in[3];
  const float* Wout   = (const float*)d_in[4];
  const float* b_out  = (const float*)d_in[5];
  const float* null_k = (const float*)d_in[6];
  const float* null_v = (const float*)d_in[7];
  const float* W_th   = (const float*)d_in[8];
  const float* b_th   = (const float*)d_in[9];
  float* out = (float*)d_out;

  char* ws = (char*)d_ws;
  size_t off = 0;
  f16* Qf = (f16*)(ws + off);   off += (size_t)BATCH * HEADS * NCHUNK * CHUNKQ * DH * 2;
  f16* Kf = (f16*)(ws + off);   off += (size_t)BATCH * HEADS * NCHUNK * JP * DH * 2;
  f16* Vf = (f16*)(ws + off);   off += (size_t)BATCH * HEADS * NCHUNK * JP * DH * 2;
  f16* At = (f16*)(ws + off);   off += (size_t)BATCH * HEADS * NCHUNK * CHUNKQ * JP * 2;
  f16* Om = (f16*)(ws + off);   off += (size_t)BATCH * 4096 * INNER * 2;
  (void)ws_size; (void)in_sizes; (void)n_in; (void)out_size;

  hipLaunchKernelGGL(proj_q_kernel,   dim3(256 * 8),  dim3(128), 0, stream, seq, Wq, Qf);
  hipLaunchKernelGGL(proj_kv_kernel,  dim3(512 * 16), dim3(128), 0, stream, ctx, Wkv, Kf, Vf);
  hipLaunchKernelGGL(fill_null_pad_kernel, dim3(BATCH * HEADS * NCHUNK), dim3(64), 0, stream,
                     null_k, null_v, Kf, Vf);
  hipLaunchKernelGGL(attn_scores_kernel, dim3(BATCH * HEADS * NCHUNK), dim3(128), 0, stream,
                     Qf, Kf, At);
  hipLaunchKernelGGL(attn_mix_av_kernel, dim3(BATCH * HEADS * NCHUNK), dim3(128), 0, stream,
                     At, Vf, W_th, b_th, Om);
  hipLaunchKernelGGL(proj_out_kernel, dim3(256 * 16), dim3(128), 0, stream, Om, Wout, b_out, out);
  hipLaunchKernelGGL(zero_first_row_kernel, dim3(BATCH), dim3(1024), 0, stream, out);
}
